// PostProcess_10428180595318
// MI455X (gfx1250) — matrix operand order repaired and checked by reading, not compiled, not run on hardware
//
#include <hip/hip_runtime.h>
#include <hip/hip_bf16.h>
#include <math.h>

typedef __attribute__((ext_vector_type(16))) _Float16 v16h;
typedef __attribute__((ext_vector_type(8)))  float    v8f;
typedef __attribute__((ext_vector_type(4)))  float    v4f;

#define B_   64
#define K_   100
#define Q_   100
#define C_   82
#define A_   118
#define AM1  117
#define KP1  101
#define NSEG 10100      // K*(K+1)
#define NT   632        // ceil(NSEG/16)
#define NCHUNK 8        // seg-tile chunks per batch (grid.y)

__device__ __forceinline__ float sigmoidf(float x){ return 1.0f/(1.0f + __expf(-x)); }

// ---------------------------------------------------------------------------
// Kernel 1: scores / labels / boxes + h_mask, o_mask
// ---------------------------------------------------------------------------
__global__ __launch_bounds__(256) void k1_scores_boxes(
    const float* __restrict__ logits,   // (B,K,C)
    const float* __restrict__ boxes,    // (B,K,4)
    const int*   __restrict__ tsizes,   // (B,2) [h,w]
    float* __restrict__ out,            // scores | labels | boxes ...
    float* __restrict__ hm,             // (B,K)
    float* __restrict__ om)             // (B,K+1)
{
  int i = blockIdx.x*blockDim.x + threadIdx.x;   // flat (b,k)
  if (i >= B_*K_) return;
  int b = i / K_, k = i % K_;

  const float* row = logits + (size_t)i * C_;
  float m = -INFINITY, m81 = -INFINITY; int idx81 = 0;
  for (int c = 0; c < C_; ++c){
    float x = row[c];
    if (x > m) m = x;
    if (c < C_-1 && x > m81){ m81 = x; idx81 = c; }   // first-occurrence argmax
  }
  float s = 0.f;
  for (int c = 0; c < C_; ++c) s += __expf(row[c] - m);
  float score = __expf(m81 - m) / s;

  out[i]           = score;          // scores
  out[B_*K_ + i]   = (float)idx81;   // labels (as float)

  const float* bb = boxes + (size_t)i*4;
  float cx = bb[0], cy = bb[1], w = bb[2], h = bb[3];
  float img_h = (float)tsizes[b*2 + 0];
  float img_w = (float)tsizes[b*2 + 1];
  float* ob = out + 2*B_*K_ + (size_t)i*4;
  ob[0] = (cx - 0.5f*w) * img_w;
  ob[1] = (cy - 0.5f*h) * img_h;
  ob[2] = (cx + 0.5f*w) * img_w;
  ob[3] = (cy + 0.5f*h) * img_h;

  hm[i] = (idx81 == 1 && score > 0.0f) ? 1.f : 0.f;
  om[b*KP1 + k] = (score > 0.0f) ? 1.f : 0.f;
  if (k == 0) om[b*KP1 + K_] = 1.f;
}

// ---------------------------------------------------------------------------
// Kernel 2: per-batch ms, argmax h/o -> f, segment-winner scan -> coeff
// coeff[q] = ms[q] * (2 if q is its segment's winner else 1)
// ---------------------------------------------------------------------------
__global__ __launch_bounds__(128) void k2_pairs(
    const float* __restrict__ actions,  // (B,Q,A)
    const float* __restrict__ hidx,     // (B,Q,K)
    const float* __restrict__ oidx,     // (B,Q,K)
    int*   __restrict__ fws,            // (B,Q)
    float* __restrict__ cws)            // (B,Q)
{
  __shared__ float msL[Q_];
  __shared__ int   fL[Q_];
  int b = blockIdx.x, t = threadIdx.x;

  if (t < Q_){
    float lastA = actions[((size_t)(b*Q_ + t))*A_ + (A_-1)];
    float ms = 1.0f - sigmoidf(lastA);
    const float* hr = hidx + (size_t)(b*Q_ + t)*K_;
    const float* orw= oidx + (size_t)(b*Q_ + t)*K_;
    float hb = -INFINITY, obv = -INFINITY; int hi = 0, oi = 0;
    for (int k = 0; k < K_; ++k){
      float hv = hr[k];  if (hv > hb){ hb = hv;  hi = k; }
      float ov = orw[k]; if (ov > obv){ obv = ov; oi = k; }
    }
    msL[t] = ms; fL[t] = hi*KP1 + oi;
  }
  __syncthreads();
  if (t < Q_){
    int seg = fL[t];
    float best = -INFINITY; int win = Q_;
    for (int q = 0; q < Q_; ++q)
      if (fL[q] == seg && msL[q] > best){ best = msL[q]; win = q; }  // strict > => min index on ties
    float coeff = msL[t] * ((win == t) ? 2.0f : 1.0f);
    fws[b*Q_ + t] = seg;
    cws[b*Q_ + t] = coeff;
  }
}

// ---------------------------------------------------------------------------
// Kernel 3: pair_score[b,a,seg] = sum_q V[a,q] * B1hot[q,seg],  via WMMA f16.
// V[a,q] = coeff[q]*sigmoid(act[b,q,a]) in f16 (LDS, stride 136 halves).
// B1hot[q,seg] = (f[q]==seg) * hm[seg/101]*om[seg%101].
// Empty seg-tiles -> NT zero stores only.
// ---------------------------------------------------------------------------
__global__ __launch_bounds__(256) void k3_pair_score(
    const float* __restrict__ actions,
    const int*   __restrict__ fws,
    const float* __restrict__ cws,
    const float* __restrict__ hm,
    const float* __restrict__ om,
    float* __restrict__ outp)           // (B,117,10100)
{
  __shared__ _Float16 Vl[128*136];      // padded A-panel, 34816 B
  __shared__ int   fL[Q_];
  __shared__ float cL[Q_];
  __shared__ float hmL[K_];
  __shared__ float omL[KP1];
  __shared__ int   flag;

  const int b = blockIdx.x;
  const int chunk = blockIdx.y;
  const int tid = threadIdx.x;

  if (tid < Q_){ fL[tid] = fws[b*Q_+tid]; cL[tid] = cws[b*Q_+tid]; hmL[tid] = hm[b*K_+tid]; }
  if (tid < KP1) omL[tid] = om[b*KP1 + tid];
  __syncthreads();

  // Build V[a][q] (zero-padded 128x128), coalesced over a, once per WG.
  {
    int a = tid & 127;
    for (int q = tid >> 7; q < 128; q += 2){
      float v = 0.f;
      if (a < AM1 && q < Q_)
        v = cL[q] * sigmoidf(actions[((size_t)(b*Q_ + q))*A_ + a]);
      Vl[a*136 + q] = (_Float16)v;
    }
  }
  __syncthreads();

  float* obase = outp + (size_t)b * AM1 * NSEG;
  const int wave = tid >> 5;      // M-tile (8 waves = 8 M-tiles of 16 rows)
  const int lane = tid & 31;
  const int n    = lane & 15;     // N / column-in-tile, also A-row-in-tile
  const int hi16 = lane >> 4;

  for (int t = chunk; t < NT; t += NCHUNK){
    const int segbase = t * 16;

    if (tid == 0) flag = 0;
    __syncthreads();
    if (tid < Q_ && (unsigned)(fL[tid] - segbase) < 16u) flag = 1;
    __syncthreads();
    const int tf = flag;
    __syncthreads();              // guard against next-iteration flag reset race

    if (tf == 0){                 // tile has no contributing query -> zeros
      if (segbase + 16 <= NSEG){
        for (int i = tid; i < AM1*4; i += 256){
          int r = i >> 2, c = i & 3;
          v4f z = {0.f, 0.f, 0.f, 0.f};
          __builtin_nontemporal_store(z, (v4f*)(obase + (size_t)r*NSEG + segbase + c*4));
        }
      } else {
        for (int i = tid; i < AM1*16; i += 256){
          int r = i >> 4, c = i & 15;
          if (segbase + c < NSEG)
            __builtin_nontemporal_store(0.f, obase + (size_t)r*NSEG + segbase + c);
        }
      }
      continue;
    }

    // per-lane column scale hm[h]*om[o] for seg = segbase + n
    float cs = 0.f;
    {
      int seg = segbase + n;
      if (seg < NSEG){ int h = seg / KP1, o = seg % KP1; cs = hmL[h] * omL[o]; }
    }

    v8f acc = {};
    #pragma unroll
    for (int ks = 0; ks < 4; ++ks){
      union { v16h h; struct { v4f lo, hi; } f; _Float16 e[16]; } Au, Bu;
      // A 16x32 f16: lane row = wave*16 + n; halves0-7 = K kb..kb+7, halves8-15 = +16
      {
        const int arow = wave*16 + n;
        const int kb   = ks*32 + hi16*8;
        const _Float16* ap = &Vl[arow*136 + kb];
        Au.f.lo = *(const v4f*)(ap);        // 16B aligned
        Au.f.hi = *(const v4f*)(ap + 16);
      }
      // B 32x16 f16: lane = column n; halves0-7 = K kb..kb+7, halves8-15 = +16 (kb = hi16*8)
      {
        const int kb = ks*32 + hi16*8;
        #pragma unroll
        for (int j = 0; j < 16; ++j){
          int q = kb + j + ((j >> 3) << 3);
          Bu.e[j] = (q < Q_ && fL[q] == segbase + n) ? (_Float16)cs : (_Float16)0.f;
        }
      }
      acc = __builtin_amdgcn_wmma_f32_16x16x32_f16(
              /*neg_a=*/false, Au.h, /*neg_b=*/false, Bu.h,
              /*c_mod=*/(short)0, acc, /*reuse_a=*/false, /*reuse_b=*/false);
    }

    // C/D 16x16 f32: VGPR r -> M = r + hi16*8, N = n
    {
      const int abase = wave*16 + hi16*8;
      const int seg = segbase + n;
      if (seg < NSEG){
        #pragma unroll
        for (int r = 0; r < 8; ++r){
          int a = abase + r;
          if (a < AM1)
            __builtin_nontemporal_store(acc[r], obase + (size_t)a*NSEG + seg);
        }
      }
    }
  }
}

// ---------------------------------------------------------------------------
extern "C" void kernel_launch(void* const* d_in, const int* in_sizes, int n_in,
                              void* d_out, int out_size, void* d_ws, size_t ws_size,
                              hipStream_t stream)
{
  const float* logits  = (const float*)d_in[0];   // pred_logits
  const float* boxes   = (const float*)d_in[1];   // pred_boxes
  const float* actions = (const float*)d_in[2];   // pred_actions
  const float* hidx    = (const float*)d_in[3];   // pred_hidx
  const float* oidx    = (const float*)d_in[4];   // pred_oidx
  const int*   tsz     = (const int*)d_in[5];     // target_sizes
  float* out = (float*)d_out;

  // workspace: f (int B*Q) | coeff (B*Q) | hm (B*K) | om (B*(K+1))  ~100 KB
  int*   fws = (int*)d_ws;
  float* cws = (float*)((char*)d_ws + (size_t)B_*Q_*sizeof(int));
  float* hmp = cws + B_*Q_;
  float* omp = hmp + B_*K_;

  k1_scores_boxes<<<(B_*K_ + 255)/256, 256, 0, stream>>>(logits, boxes, tsz, out, hmp, omp);
  k2_pairs<<<B_, 128, 0, stream>>>(actions, hidx, oidx, fws, cws);

  dim3 g3(B_, NCHUNK);
  k3_pair_score<<<g3, 256, 0, stream>>>(actions, fws, cws, hmp, omp,
                                        out + (size_t)B_*K_*6 /* 38400 */);
}
